// DeepGCN_Aux_90821378441627
// MI455X (gfx1250) — matrix operand order repaired and checked by reading, not compiled, not run on hardware
//
#include <hip/hip_runtime.h>

// ---------------- problem constants ----------------
#define B_   8
#define T_   8
#define N_   512
#define CPC_ 7
#define FS_  8
#define K_   9
#define CH_  64
#define NL_  7
#define NC_  49
#define G_   (B_*T_)            // 64 groups
#define NT_  (G_*N_)            // 32768 nodes
#define NE_  ((size_t)NT_*K_)   // 294912 edges
#define EPS_ 1e-5f

typedef __attribute__((ext_vector_type(16))) _Float16 v16h;
typedef __attribute__((ext_vector_type(8)))  _Float16 v8h;
typedef __attribute__((ext_vector_type(8)))  float    v8f;

// ---------------- elementwise helpers ----------------
__global__ void k_f32_to_f16(const float* __restrict__ s, _Float16* __restrict__ d, size_t n) {
  size_t i = (size_t)blockIdx.x * blockDim.x + threadIdx.x;
  if (i < n) d[i] = (_Float16)s[i];
}
__global__ void k_fill_f32(float* __restrict__ p, size_t n, float v) {
  size_t i = (size_t)blockIdx.x * blockDim.x + threadIdx.x;
  if (i < n) p[i] = v;
}

// Build padded f16 input features: NT x 32 (cols 0..6 = point cloud, 7..14 = frame sig, rest 0)
__global__ void k_build_x0(const float* __restrict__ pc, const float* __restrict__ fsig,
                           _Float16* __restrict__ x0) {
  size_t i = (size_t)blockIdx.x * blockDim.x + threadIdx.x;
  size_t n = (size_t)NT_ * 32;
  if (i >= n) return;
  int row = (int)(i >> 5);
  int c   = (int)(i & 31);
  float v = 0.0f;
  if (c < CPC_)            v = pc[(size_t)row * CPC_ + c];
  else if (c < CPC_ + FS_) v = fsig[(size_t)(row / N_) * FS_ + (c - CPC_)];
  x0[i] = (_Float16)v;
}

// Repack fp32 weight [Krows x Ncols] into WMMA-B-fragment-native f16 layout:
//   Wp[ ((nTile*32 + lane)*KT + kt)*16 + i ]  where k = kt*32 + (lane>>4)*16 + i,
//   n = nTile*16 + (lane&15).  Rows k >= Krows are zero (handles stem 15->32 pad).
__global__ void k_pack_w(const float* __restrict__ W, _Float16* __restrict__ Wp,
                         int Krows, int KT, int Ncols) {
  size_t total = (size_t)KT * 32 * Ncols;   // == Kdim*Ncols halves
  size_t idx = (size_t)blockIdx.x * blockDim.x + threadIdx.x;
  if (idx >= total) return;
  int i    = (int)(idx & 15);
  size_t r = idx >> 4;
  int kt   = (int)(r % (size_t)KT);  r /= (size_t)KT;
  int lane = (int)(r & 31);
  int t    = (int)(r >> 5);
  int k = kt * 32 + (lane >> 4) * 16 + i;
  int n = t * 16 + (lane & 15);
  float v = (k < Krows) ? W[(size_t)k * Ncols + n] : 0.0f;
  Wp[idx] = (_Float16)v;
}

// ---------------- KNN (top-9 by squared distance, stable ties) ----------------
__global__ void k_knn(const float* __restrict__ pc, int* __restrict__ srcIdx) {
  __shared__ float sx[N_], sy[N_], sz[N_];
  int g = blockIdx.x;          // group 0..63
  int i = threadIdx.x;         // point 0..511
  const float* base = pc + (size_t)g * N_ * CPC_;
  sx[i] = base[(size_t)i * CPC_ + 0];
  sy[i] = base[(size_t)i * CPC_ + 1];
  sz[i] = base[(size_t)i * CPC_ + 2];
  __syncthreads();
  float xi = sx[i], yi = sy[i], zi = sz[i];
  float bd[K_]; int bi[K_];
#pragma unroll
  for (int k = 0; k < K_; ++k) { bd[k] = 3.4e38f; bi[k] = 0; }
  for (int j = 0; j < N_; ++j) {
    float dx = sx[j] - xi, dy = sy[j] - yi, dz = sz[j] - zi;
    float d2 = dx * dx + dy * dy + dz * dz;
    if (j == i) d2 += 1e10f;   // matches +eye*1e10
    if (d2 < bd[K_ - 1]) {     // strict < => stable tie-break (lower index first)
      int k = K_ - 1;
      while (k > 0 && bd[k - 1] > d2) { bd[k] = bd[k - 1]; bi[k] = bi[k - 1]; --k; }
      bd[k] = d2; bi[k] = j;
    }
  }
  size_t e = ((size_t)(g * N_ + i)) * K_;
#pragma unroll
  for (int k = 0; k < K_; ++k) srcIdx[e + k] = g * N_ + bi[k];
}

// ---------------- WMMA GEMM: one wave computes a 64x64 output tile ----------------
// MODE 0: A dense f16 [M x KTILES*32].
// MODE 1: edge-gather: row r = concat(feat[src[r]], feat[r/K]), cin = KTILES*16.
// Wp: fragment-packed weights (see k_pack_w). Out fp32 [M x Ncols] = A@W + bias.
// grid = (M/64, Ncols/64), block = 32 (one wave). Fully unrolled: per kt,
// 4 B-frag loads + 4 A-frag loads feed 16 v_wmma (each fragment reused 4x).
template <int KTILES, int MODE>
__global__ __launch_bounds__(32)
void k_wmma_gemm(const _Float16* __restrict__ A,
                 const _Float16* __restrict__ Wp,
                 const float* __restrict__ bias,
                 const int* __restrict__ srcIdx,
                 float* __restrict__ Cout,
                 int Ncols) {
  constexpr int KDIM = KTILES * 32;
  constexpr int CIN  = KTILES * 16;   // mode 1 only
  int lane    = threadIdx.x & 31;
  int half_id = lane >> 4;
  int l16     = lane & 15;
  int m0 = blockIdx.x * 64;
  int n0 = blockIdx.y * 64;

  const _Float16* arow[4];
  const _Float16* srowp[4];
  const _Float16* crowp[4];
#pragma unroll
  for (int m = 0; m < 4; ++m) {
    int r = m0 + m * 16 + l16;
    if (MODE == 0) {
      arow[m] = A + (size_t)r * KDIM;
    } else {
      srowp[m] = A + (size_t)srcIdx[r] * CIN;
      crowp[m] = A + (size_t)(r / K_) * CIN;
    }
  }

  v8f acc[4][4];
#pragma unroll
  for (int m = 0; m < 4; ++m)
#pragma unroll
    for (int n = 0; n < 4; ++n) acc[m][n] = (v8f){};

#pragma unroll
  for (int kt = 0; kt < KTILES; ++kt) {
    // B fragments for the 4 N-tiles: contiguous 32B per lane (packed layout)
    v16h b[4];
#pragma unroll
    for (int n = 0; n < 4; ++n) {
      const _Float16* wp = Wp + ((size_t)((n0 >> 4) + n) * 32 + lane) * (KTILES * 16)
                              + (size_t)kt * 16;
      v8h w0 = *(const v8h*)wp;
      v8h w1 = *(const v8h*)(wp + 8);
#pragma unroll
      for (int i = 0; i < 8; ++i) { b[n][i] = w0[i]; b[n][8 + i] = w1[i]; }
    }
#pragma unroll
    for (int m = 0; m < 4; ++m) {
      int k0 = kt * 32 + half_id * 16;
      int k1 = kt * 32 + half_id * 16 + 8;
      v8h lo, hi;
      if (MODE == 0) {
        lo = *(const v8h*)(arow[m] + k0);
        hi = *(const v8h*)(arow[m] + k1);
      } else {
        // K-tile halves never straddle the concat boundary (CIN % 16 == 0),
        // so these conditions are compile-time constants after unrolling.
        const _Float16* p0 = (k0 < CIN)      ? srowp[m] + k0 : crowp[m] + (k0 - CIN);
        const _Float16* p1 = (k1 < CIN) ? srowp[m] + k1 : crowp[m] + (k1 - CIN);
        lo = *(const v8h*)p0;
        hi = *(const v8h*)p1;
      }
      v16h a;
#pragma unroll
      for (int i = 0; i < 8; ++i) { a[i] = lo[i]; a[8 + i] = hi[i]; }
#pragma unroll
      for (int n = 0; n < 4; ++n)
        acc[m][n] = __builtin_amdgcn_wmma_f32_16x16x32_f16(
            false, a, false, b[n], (short)0, acc[m][n], false, false);
    }
  }

  // C/D layout: VGPR r -> (M = r + 8*half_id, N = lane&15)
#pragma unroll
  for (int n = 0; n < 4; ++n) {
    int col = n0 + n * 16 + l16;
    float bv = bias[col];
#pragma unroll
    for (int m = 0; m < 4; ++m) {
      int rbase = m0 + m * 16 + half_id * 8;
#pragma unroll
      for (int r = 0; r < 8; ++r)
        Cout[(size_t)(rbase + r) * Ncols + col] = acc[m][n][r] + bv;
    }
  }
}

// ---------------- BatchNorm statistics (deterministic two-stage) ----------------
#define STAT_ROWS 1024
__global__ void k_bn_partial(const float* __restrict__ x, int M, int C,
                             float* __restrict__ psum, float* __restrict__ psq) {
  __shared__ float ssum[256], ssq[256];
  int t = threadIdx.x;
  int c = t % C;                 // C = 64 or 128, divides 256
  int r0 = t / C;
  int stride = 256 / C;
  int base = blockIdx.x * STAT_ROWS;
  int end = base + STAT_ROWS; if (end > M) end = M;
  float s = 0.f, q = 0.f;
  for (int r = base + r0; r < end; r += stride) {
    float v = x[(size_t)r * C + c];
    s += v; q += v * v;
  }
  ssum[t] = s; ssq[t] = q;
  __syncthreads();
  if (t < C) {
    float S = 0.f, Q = 0.f;
    for (int i = 0; i < stride; ++i) { S += ssum[t + i * C]; Q += ssq[t + i * C]; }
    psum[(size_t)blockIdx.x * C + t] = S;
    psq [(size_t)blockIdx.x * C + t] = Q;
  }
}
__global__ void k_bn_finalize(const float* __restrict__ psum, const float* __restrict__ psq,
                              int P, int C, float invM,
                              float* __restrict__ mean, float* __restrict__ istd) {
  int c = threadIdx.x;
  if (c >= C) return;
  float S = 0.f, Q = 0.f;
  for (int i = 0; i < P; ++i) { S += psum[(size_t)i * C + c]; Q += psq[(size_t)i * C + c]; }
  float mu = S * invM;
  float var = Q * invM - mu * mu;
  if (var < 0.f) var = 0.f;
  mean[c] = mu;
  istd[c] = rsqrtf(var + EPS_);
}

// y = relu( (x-mu)*istd*g + be [+ identity] ); write f32 and/or f16
__global__ void k_bn_apply(const float* __restrict__ x,
                           const float* __restrict__ mean, const float* __restrict__ istd,
                           const float* __restrict__ g, const float* __restrict__ be,
                           const float* __restrict__ identity,
                           float* __restrict__ outf, _Float16* __restrict__ outh,
                           size_t n, int C) {
  size_t i = (size_t)blockIdx.x * blockDim.x + threadIdx.x;
  if (i >= n) return;
  int c = (int)(i % (size_t)C);
  float v = (x[i] - mean[c]) * istd[c] * g[c] + be[c];
  if (identity) v += identity[i];
  v = fmaxf(v, 0.0f);
  if (outf) outf[i] = v;
  if (outh) outh[i] = (_Float16)v;
}

// segment_max by src; h >= 0 post-ReLU so uint order == float order; agg pre-zeroed.
__global__ void k_seg_max(const float* __restrict__ h, const int* __restrict__ srcIdx,
                          unsigned int* __restrict__ agg, size_t n, int C) {
  size_t i = (size_t)blockIdx.x * blockDim.x + threadIdx.x;
  if (i >= n) return;
  size_t e = i / (size_t)C;
  int c = (int)(i % (size_t)C);
  unsigned int v = __float_as_uint(h[i]);
  atomicMax(&agg[(size_t)srcIdx[e] * C + c], v);
}

// ---------------- pooling + head ----------------
__global__ void k_pool_max(const float* __restrict__ h, float* __restrict__ pooled, int C) {
  int g = blockIdx.x, c = threadIdx.x;
  float m = -3.4e38f;
  for (int n = 0; n < N_; ++n) m = fmaxf(m, h[(size_t)(g * N_ + n) * C + c]);
  pooled[(size_t)g * C + c] = m;
}
__global__ void k_pool_mean(const float* __restrict__ pooled, float* __restrict__ feat, int C) {
  int b = blockIdx.x, c = threadIdx.x;
  float s = 0.f;
  for (int t = 0; t < T_; ++t) s += pooled[(size_t)(b * T_ + t) * C + c];
  feat[(size_t)b * C + c] = s / (float)T_;
}
__global__ void k_linear_small(const float* __restrict__ in, const float* __restrict__ W,
                               const float* __restrict__ bias, float* __restrict__ out,
                               int Mr, int Kd, int Nc, int relu) {
  int idx = blockIdx.x * blockDim.x + threadIdx.x;
  if (idx >= Mr * Nc) return;
  int m = idx / Nc, n = idx % Nc;
  float s = bias[n];
  for (int k = 0; k < Kd; ++k) s += in[(size_t)m * Kd + k] * W[(size_t)k * Nc + n];
  out[idx] = relu ? fmaxf(s, 0.f) : s;
}

// ---------------- host orchestration ----------------
extern "C" void kernel_launch(void* const* d_in, const int* in_sizes, int n_in,
                              void* d_out, int out_size, void* d_ws, size_t ws_size,
                              hipStream_t stream) {
  (void)in_sizes; (void)n_in; (void)out_size; (void)ws_size;
  const float* pc   = (const float*)d_in[0];
  const float* fsig = (const float*)d_in[1];

  // params in setup_inputs() insertion order
  int pi = 2;
  auto nxt = [&]() { return (const float*)d_in[pi++]; };
  const float *st_l0W = nxt(), *st_l0b = nxt(), *st_bg = nxt(), *st_bb = nxt(),
              *st_l1W = nxt(), *st_l1b = nxt();
  struct LP {
    const float *e0W, *e0b, *bn0g, *bn0b, *e1W, *e1b, *bn1g, *bn1b, *bng, *bnb, *resW, *resb;
    int cin, cout;
  } lp[NL_];
  {
    int cin = CH_;
    for (int i = 0; i < NL_; ++i) {
      int cout = (i == NL_ / 3) ? CH_ * 2 : cin;
      lp[i].e0W = nxt(); lp[i].e0b = nxt(); lp[i].bn0g = nxt(); lp[i].bn0b = nxt();
      lp[i].e1W = nxt(); lp[i].e1b = nxt(); lp[i].bn1g = nxt(); lp[i].bn1b = nxt();
      lp[i].bng = nxt(); lp[i].bnb = nxt();
      lp[i].resW = nullptr; lp[i].resb = nullptr;
      if (cin != cout) { lp[i].resW = nxt(); lp[i].resb = nxt(); }
      lp[i].cin = cin; lp[i].cout = cout;
      cin = cout;
    }
  }
  const float *hd_l0W = nxt(), *hd_l0b = nxt(), *hd_l1W = nxt(), *hd_l1b = nxt(),
              *hd_l2W = nxt(), *hd_l2b = nxt();

  // ---- workspace carve-up ----
  char* wsb = (char*)d_ws;
  size_t off = 0;
  auto alloc = [&](size_t bytes) { size_t o = off; off = (off + bytes + 255) & ~(size_t)255; return o; };
  _Float16* x0h    = (_Float16*)(wsb + alloc((size_t)NT_ * 32 * 2));
  int*      srcIdx = (int*)     (wsb + alloc(NE_ * 4));
  float*    hAf    = (float*)   (wsb + alloc((size_t)NT_ * 128 * 4));
  _Float16* hAh    = (_Float16*)(wsb + alloc((size_t)NT_ * 128 * 2));
  float*    hBf    = (float*)   (wsb + alloc((size_t)NT_ * 128 * 4));
  _Float16* hBh    = (_Float16*)(wsb + alloc((size_t)NT_ * 128 * 2));
  float*    ef     = (float*)   (wsb + alloc(NE_ * 128 * 4));   // edge fp32 (also stem/res scratch)
  _Float16* eh     = (_Float16*)(wsb + alloc(NE_ * 128 * 2));   // edge f16
  int maxP = (int)((NE_ + STAT_ROWS - 1) / STAT_ROWS);          // 288
  float*    psum   = (float*)   (wsb + alloc((size_t)maxP * 128 * 4));
  float*    psq    = (float*)   (wsb + alloc((size_t)maxP * 128 * 4));
  float*    meanb  = (float*)   (wsb + alloc(128 * 4));
  float*    istdb  = (float*)   (wsb + alloc(128 * 4));
  _Float16* warena = (_Float16*)(wsb + alloc(1u << 20));
  float*    pooled = (float*)   (wsb + alloc((size_t)G_ * 128 * 4));
  float*    feat   = (float*)   (wsb + alloc((size_t)B_ * 128 * 4));
  float*    th1    = (float*)   (wsb + alloc((size_t)B_ * 256 * 4));
  float*    th2    = (float*)   (wsb + alloc((size_t)B_ * 128 * 4));

  auto blk = [](size_t n) { return (unsigned)((n + 255) / 256); };

  // pack weights (fp32 -> fragment-native f16); Krows < KT*32 zero-pads (stem)
  _Float16* wcur = warena;
  auto packw = [&](const float* W, int Krows, int KT, int Nc) {
    _Float16* slot = wcur;
    size_t nh = (size_t)KT * 32 * Nc;
    wcur += (nh + 127) & ~(size_t)127;
    k_pack_w<<<blk(nh), 256, 0, stream>>>(W, slot, Krows, KT, Nc);
    return slot;
  };
  _Float16* w_st0 = packw(st_l0W, CPC_ + FS_, 1, 64);   // 15 real rows, pad to 32
  _Float16* w_st1 = packw(st_l1W, 64, 2, 64);
  _Float16 *w_e0[NL_], *w_e1[NL_], *w_res = nullptr;
  for (int i = 0; i < NL_; ++i) {
    w_e0[i] = packw(lp[i].e0W, 2 * lp[i].cin, (2 * lp[i].cin) / 32, lp[i].cout);
    w_e1[i] = packw(lp[i].e1W, lp[i].cout, lp[i].cout / 32, lp[i].cout);
    if (lp[i].resW) w_res = packw(lp[i].resW, lp[i].cin, lp[i].cin / 32, lp[i].cout);
  }

  auto gemm = [&](const _Float16* A, const _Float16* Wp, const float* bias, const int* src,
                  float* out, size_t M, int Kd, int Nc, int mode) {
    dim3 grid((unsigned)(M / 64), (unsigned)(Nc / 64));
    int kt = Kd / 32;
    if (mode == 0) {
      if      (kt == 1) k_wmma_gemm<1, 0><<<grid, 32, 0, stream>>>(A, Wp, bias, src, out, Nc);
      else if (kt == 2) k_wmma_gemm<2, 0><<<grid, 32, 0, stream>>>(A, Wp, bias, src, out, Nc);
      else              k_wmma_gemm<4, 0><<<grid, 32, 0, stream>>>(A, Wp, bias, src, out, Nc);
    } else {
      if (kt == 4)      k_wmma_gemm<4, 1><<<grid, 32, 0, stream>>>(A, Wp, bias, src, out, Nc);
      else              k_wmma_gemm<8, 1><<<grid, 32, 0, stream>>>(A, Wp, bias, src, out, Nc);
    }
  };
  auto bn_stats = [&](const float* x, size_t M, int C) {
    int P = (int)((M + STAT_ROWS - 1) / STAT_ROWS);
    k_bn_partial<<<(unsigned)P, 256, 0, stream>>>(x, (int)M, C, psum, psq);
    k_bn_finalize<<<1, (unsigned)C, 0, stream>>>(psum, psq, P, C, 1.0f / (float)M, meanb, istdb);
  };
  auto bn_apply = [&](const float* x, const float* g, const float* be, const float* identity,
                      float* outf, _Float16* outh, size_t M, int C) {
    size_t n = M * (size_t)C;
    k_bn_apply<<<blk(n), 256, 0, stream>>>(x, meanb, istdb, g, be, identity, outf, outh, n, C);
  };

  // ---- 1. inputs + KNN ----
  k_build_x0<<<blk((size_t)NT_ * 32), 256, 0, stream>>>(pc, fsig, x0h);
  k_knn<<<G_, N_, 0, stream>>>(pc, srcIdx);

  // ---- 2. stem ----
  gemm(x0h, w_st0, st_l0b, nullptr, ef, NT_, 32, 64, 0);          // 15(pad32)->64
  bn_stats(ef, NT_, 64);
  bn_apply(ef, st_bg, st_bb, nullptr, nullptr, eh, NT_, 64);      // relu -> f16
  gemm(eh, w_st1, st_l1b, nullptr, hAf, NT_, 64, 64, 0);          // 64->64 (no act)
  k_f32_to_f16<<<blk((size_t)NT_ * 64), 256, 0, stream>>>(hAf, hAh, (size_t)NT_ * 64);

  // ---- 3. GCN layers ----
  for (int i = 0; i < NL_; ++i) {
    int cin = lp[i].cin, cout = lp[i].cout;
    // edge MLP stage 1: gathered concat(x[src],x[ctr]) @ e0.W
    gemm(hAh, w_e0[i], lp[i].e0b, srcIdx, ef, NE_, 2 * cin, cout, 1);
    bn_stats(ef, NE_, cout);
    bn_apply(ef, lp[i].bn0g, lp[i].bn0b, nullptr, nullptr, eh, NE_, cout);
    // stage 2
    gemm(eh, w_e1[i], lp[i].e1b, nullptr, ef, NE_, cout, cout, 0);
    bn_stats(ef, NE_, cout);
    bn_apply(ef, lp[i].bn1g, lp[i].bn1b, nullptr, ef, nullptr, NE_, cout); // relu, f32 in-place
    // segment max by src into hBf (zero == empty-segment fixup since h>=0)
    k_fill_f32<<<blk((size_t)NT_ * cout), 256, 0, stream>>>(hBf, (size_t)NT_ * cout, 0.0f);
    k_seg_max<<<blk(NE_ * (size_t)cout), 256, 0, stream>>>(ef, srcIdx, (unsigned int*)hBf,
                                                           NE_ * (size_t)cout, cout);
    // residual path
    const float* identity = hAf;
    if (lp[i].resW) {
      gemm(hAh, w_res, lp[i].resb, nullptr, ef, NT_, cin, cout, 0); // reuse ef as res buffer
      identity = ef;
    }
    // out = relu(bn(agg) + identity)
    bn_stats(hBf, NT_, cout);
    bn_apply(hBf, lp[i].bng, lp[i].bnb, identity, hBf, hBh, NT_, cout);
    // swap ping-pong
    { float* tf = hAf; hAf = hBf; hBf = tf; }
    { _Float16* th = hAh; hAh = hBh; hBh = th; }
  }

  // ---- 4. pooling + head ----
  int C = lp[NL_ - 1].cout;                                       // 128
  k_pool_max<<<G_, (unsigned)C, 0, stream>>>(hAf, pooled, C);
  k_pool_mean<<<B_, (unsigned)C, 0, stream>>>(pooled, feat, C);
  k_linear_small<<<blk((size_t)B_ * 256), 256, 0, stream>>>(feat, hd_l0W, hd_l0b, th1, B_, 128, 256, 1);
  k_linear_small<<<blk((size_t)B_ * 128), 256, 0, stream>>>(th1, hd_l1W, hd_l1b, th2, B_, 256, 128, 1);
  k_linear_small<<<blk((size_t)B_ * NC_), 256, 0, stream>>>(th2, hd_l2W, hd_l2b, (float*)d_out, B_, 128, NC_, 0);
}
